// DistSAGE_42167988912731
// MI455X (gfx1250) — matrix operand order repaired and checked by reading, not compiled, not run on hardware
//
#include <hip/hip_runtime.h>

#define DEV __device__ __forceinline__

typedef __attribute__((ext_vector_type(16))) __bf16          v16bf;
typedef __attribute__((ext_vector_type(8)))  float           v8f;
typedef __attribute__((ext_vector_type(16))) unsigned short  us16v;
typedef __attribute__((ext_vector_type(8)))  unsigned short  us8v;
typedef __attribute__((ext_vector_type(4)))  unsigned int    u32x4;
typedef __attribute__((ext_vector_type(8)))  int             i32x8;
typedef __attribute__((ext_vector_type(4)))  int             i32x4;

enum : int { cN0 = 524288, cN1 = 131072, cN2 = 16384, cN3 = 8192,
             cIN = 128, cHID = 256, cOUT = 47, cOUTP = 64, cFAN = 16 };

// ---------- helpers ----------
DEV unsigned short f2bf(float f) {          // native cvt if available
  return __builtin_bit_cast(unsigned short, static_cast<__bf16>(f));
}
DEV float bf2f(unsigned short b) {
  return __builtin_bit_cast(float, (unsigned)b << 16);
}
DEV unsigned pcg_hash(unsigned v) {         // deterministic dropout mask
  v = v * 747796405u + 2891336453u;
  unsigned w = ((v >> ((v >> 28u) + 4u)) ^ v) * 277803737u;
  return (w >> 22u) ^ w;
}

// A-fragment (16x32 bf16): lane holds row (lane&15), K-halves (lane>>4)*8 + {0..7,16..23}.
DEV v16bf frag16(const unsigned short* p) {
  us8v lo = *(const us8v*)p;
  us8v hi = *(const us8v*)(p + 16);
  us16v a = __builtin_shufflevector(lo, hi, 0,1,2,3,4,5,6,7,8,9,10,11,12,13,14,15);
  return __builtin_bit_cast(v16bf, a);
}
// B-fragment (32x16 bf16) from pre-transposed weight WT[N][K]:
// lane holds column (lane&15), K-halves k0 + (lane>>4)*8 + {0..7,16..23}
DEV v16bf load_b(const unsigned short* __restrict__ wT, int col, int K, int k0, int kb) {
  const unsigned short* p = wT + (size_t)col * K + k0 + kb;
  us8v lo = *(const us8v*)(p);
  us8v hi = *(const us8v*)(p + 16);
  us16v b = __builtin_shufflevector(lo, hi, 0,1,2,3,4,5,6,7,8,9,10,11,12,13,14,15);
  return __builtin_bit_cast(v16bf, b);
}

// ---------- kernel 1: weight convert + transpose  w[K][N] f32 -> wT[Npad][K] bf16 ----------
__global__ void w_transpose_bf16(const float* __restrict__ w, unsigned short* __restrict__ wT,
                                 int K, int N, int Npad) {
  int idx = blockIdx.x * blockDim.x + threadIdx.x;
  int total = Npad * K;
  if (idx >= total) return;
  int n = idx / K, k = idx - n * K;
  float v = (n < N) ? w[(size_t)k * N + n] : 0.0f;
  wT[(size_t)n * K + k] = f2bf(v);
}

// ---------- kernel 2: edge-parallel segment-sum (one wave per edge, grid-stride) ----------
template <bool SRCBF>
__global__ void sage_aggregate(const void* __restrict__ feat, const int* __restrict__ src,
                               const int* __restrict__ dst, float* __restrict__ agg,
                               float* __restrict__ deg, int E, int K) {
  const int lane = threadIdx.x & 31;
  const int wid  = (blockIdx.x * blockDim.x + threadIdx.x) >> 5;
  const int nwav = (gridDim.x * blockDim.x) >> 5;
  for (int e = wid; e < E; e += nwav) {
    const int s = src[e];
    const int d = dst[e];
    int e2 = e + nwav;                      // prefetch next edge's source row
    if (e2 < E) {
      int s2 = src[e2];
      if (SRCBF) __builtin_prefetch((const unsigned short*)feat + (size_t)s2 * K, 0, 1);
      else       __builtin_prefetch((const float*)feat + (size_t)s2 * K, 0, 1);
    }
    float* ap = agg + (size_t)d * K;
    if (SRCBF) {
      const unsigned short* sp = (const unsigned short*)feat + (size_t)s * K;
#pragma unroll 4
      for (int f = lane; f < K; f += 32)
        __hip_atomic_fetch_add(&ap[f], bf2f(sp[f]), __ATOMIC_RELAXED, __HIP_MEMORY_SCOPE_AGENT);
    } else {
      const float* sp = (const float*)feat + (size_t)s * K;
#pragma unroll 4
      for (int f = lane; f < K; f += 32)
        __hip_atomic_fetch_add(&ap[f], sp[f], __ATOMIC_RELAXED, __HIP_MEMORY_SCOPE_AGENT);
    }
    if (lane == 0)
      __hip_atomic_fetch_add(&deg[d], 1.0f, __ATOMIC_RELAXED, __HIP_MEMORY_SCOPE_AGENT);
  }
}

// ---------- kernel 3: fused SAGE layer GEMM via V_WMMA_F32_16X16X32_BF16 ----------
// Block = 256 threads (8 waves); block tile = 16 rows x (NTL*16) cols, grid.y = 1.
// Self panel: bf16 layers DMA'd into LDS by the Tensor Data Mover (tensor_load_to_lds +
// s_wait_tensorcnt); f32 layer-0 panel converted once per block. Neighbor panel normalized
// + converted once per block. All A fragments come from LDS (ds_load_b128).
// Wave w computes n-tiles w and w+8 (2 accumulators, 4 WMMAs per K-step).
template <bool SRC_BF16, bool RELU_DROP, int KT, int NTL>
__global__ void sage_gemm(const void* __restrict__ srcfeat,
                          const float* __restrict__ agg,
                          const float* __restrict__ deg,
                          const unsigned short* __restrict__ wsT,
                          const unsigned short* __restrict__ wnT,
                          const float* __restrict__ bias,
                          void* __restrict__ out,
                          int NOUT, unsigned seed) {
  __shared__ __align__(16) unsigned short sN[16 * KT];   // neigh panel (bf16)
  __shared__ __align__(16) unsigned short sS[16 * KT];   // self panel (bf16)
  __shared__ float sR[16];

  const int tid  = threadIdx.x;
  const int lane = tid & 31;
  const int wave = (tid >> 5) & 7;     // &7 -> value range known, guards fold for NTL==16
  const int m0   = blockIdx.x * 16;

  if (tid < 16) sR[tid] = 1.0f / fmaxf(deg[m0 + tid], 1.0f);

  // ---- self panel: TDM DMA (bf16 layers) ----
  if constexpr (SRC_BF16) {
    if (wave == 0) {
      const unsigned long long ga =
          (unsigned long long)(const void*)((const unsigned short*)srcfeat + (size_t)m0 * KT);
      const unsigned ldsa = (unsigned)(unsigned long long)(void*)sS;
      u32x4 g0 = (u32x4)0u;
      i32x8 g1 = (i32x8)0;
      i32x4 g2 = (i32x4)0, g3 = (i32x4)0;
      g0[0] = 1u;                                   // count = 1 valid descriptor
      g0[1] = ldsa;                                 // lds_addr (bytes)
      g0[2] = (unsigned)ga;                         // global_addr[31:0]
      g0[3] = (unsigned)(ga >> 32) | 0x80000000u;   // global_addr[56:32] | type=2
      g1[0] = (int)(1u << 16);                      // data_size = 1 (2 bytes), mask=0
      g1[1] = (int)(((unsigned)KT & 0xFFFFu) << 16);// tensor_dim0[15:0] = KT
      g1[2] = (int)(16u << 16);                     // tensor_dim0[31:16]=0 | tensor_dim1 lo = 16
      g1[3] = (int)((unsigned)KT << 16);            // tensor_dim1 hi = 0 | tile_dim0 = KT
      g1[4] = (int)16u;                             // tile_dim1 = 16, tile_dim2 = 0
      g1[5] = (int)(unsigned)KT;                    // tensor_dim0_stride[31:0] = KT
      g1[6] = 0; g1[7] = 0;                         // stride hi / dim1_stride = 0
      __builtin_amdgcn_tensor_load_to_lds(g0, g1, g2, g3, (i32x8)0, 0);
      __builtin_amdgcn_s_wait_tensorcnt(0);
    }
  }
  __syncthreads();

  // ---- stage neigh (and f32 self) panels into LDS, converting f32 -> bf16 once ----
  constexpr int KSH = (KT == 128) ? 7 : 8;
#pragma unroll 4
  for (int i = tid; i < 16 * KT; i += 256) {
    const int r = i >> KSH;
    const int k = i & (KT - 1);
    sN[i] = f2bf(agg[(size_t)(m0 + r) * KT + k] * sR[r]);
    if (!SRC_BF16)
      sS[i] = f2bf(((const float*)srcfeat)[(size_t)(m0 + r) * KT + k]);
  }
  __syncthreads();

  const int rowL = lane & 15;
  const int colL = lane & 15;
  const int koff = (lane >> 4) * 8;    // A K-half select
  const int kb   = (lane >> 4) * 8;    // B K-half select
  const int nt0  = wave;
  const int nt1  = wave + 8;

  v8f c0 = (v8f)0.0f, c1 = (v8f)0.0f;
#pragma unroll
  for (int k0 = 0; k0 < KT; k0 += 32) {
    v16bf as = frag16(&sS[rowL * KT + k0 + koff]);
    v16bf an = frag16(&sN[rowL * KT + k0 + koff]);
    if (nt0 < NTL) {                   // always true for NTL==16 (wave in [0,7])
      v16bf bs = load_b(wsT, nt0 * 16 + colL, KT, k0, kb);
      v16bf bn = load_b(wnT, nt0 * 16 + colL, KT, k0, kb);
      c0 = __builtin_amdgcn_wmma_f32_16x16x32_bf16(false, as, false, bs, (short)0, c0, false, false);
      c0 = __builtin_amdgcn_wmma_f32_16x16x32_bf16(false, an, false, bn, (short)0, c0, false, false);
    }
    if (nt1 < NTL) {                   // always true for NTL==16
      v16bf bs = load_b(wsT, nt1 * 16 + colL, KT, k0, kb);
      v16bf bn = load_b(wnT, nt1 * 16 + colL, KT, k0, kb);
      c1 = __builtin_amdgcn_wmma_f32_16x16x32_bf16(false, as, false, bs, (short)0, c1, false, false);
      c1 = __builtin_amdgcn_wmma_f32_16x16x32_bf16(false, an, false, bn, (short)0, c1, false, false);
    }
  }

  // ---- epilogue: C/D layout: lane holds column (lane&15); VGPR r holds row 8*(lane>>4)+r ----
  const int N = NTL * 16;
  auto epilogue = [&](const v8f& c, int nt) {
    if (nt >= NTL) return;
    const int nn = nt * 16 + colL;
    float bv;
    if (RELU_DROP) bv = bias[nn];
    else           bv = (nn < NOUT) ? bias[nn] : 0.0f;
#pragma unroll
    for (int r = 0; r < 8; ++r) {
      const int mm = m0 + ((lane >> 4) * 8) + r;
      float v = c[r] + bv;
      if (RELU_DROP) {
        v = fmaxf(v, 0.0f);
        unsigned h = pcg_hash(((unsigned)mm * (unsigned)N + (unsigned)nn) ^ seed);
        v = (h & 1u) ? (v * 2.0f) : 0.0f;   // p = 0.5, scale 1/(1-p) = 2
        ((unsigned short*)out)[(size_t)mm * N + nn] = f2bf(v);
      } else {
        if (nn < NOUT)
          ((float*)out)[(size_t)mm * NOUT + nn] = v;
      }
    }
  };
  epilogue(c0, nt0);
  epilogue(c1, nt1);
}

// ---------- host ----------
extern "C" void kernel_launch(void* const* d_in, const int* in_sizes, int n_in,
                              void* d_out, int out_size, void* d_ws, size_t ws_size,
                              hipStream_t stream) {
  const float* x    = (const float*)d_in[0];
  const int*   src0 = (const int*)d_in[1];
  const int*   dst0 = (const int*)d_in[2];
  const int*   src1 = (const int*)d_in[3];
  const int*   dst1 = (const int*)d_in[4];
  const int*   src2 = (const int*)d_in[5];
  const int*   dst2 = (const int*)d_in[6];
  const float* ws0  = (const float*)d_in[7];
  const float* wn0  = (const float*)d_in[8];
  const float* b0   = (const float*)d_in[9];
  const float* ws1  = (const float*)d_in[10];
  const float* wn1  = (const float*)d_in[11];
  const float* b1   = (const float*)d_in[12];
  const float* ws2  = (const float*)d_in[13];
  const float* wn2  = (const float*)d_in[14];
  const float* b2   = (const float*)d_in[15];

  char* base = (char*)d_ws;
  size_t off = 0;
  auto alloc = [&](size_t bytes) -> char* {
    char* p = base + off;
    off = (off + bytes + 255) & ~(size_t)255;
    return p;
  };

  float*          agg0 = (float*)alloc((size_t)cN1 * cIN  * 4);
  float*          deg0 = (float*)alloc((size_t)cN1 * 4);
  unsigned short* h1   = (unsigned short*)alloc((size_t)cN1 * cHID * 2);
  float*          agg1 = (float*)alloc((size_t)cN2 * cHID * 4);
  float*          deg1 = (float*)alloc((size_t)cN2 * 4);
  unsigned short* h2   = (unsigned short*)alloc((size_t)cN2 * cHID * 2);
  float*          agg2 = (float*)alloc((size_t)cN3 * cHID * 4);
  float*          deg2 = (float*)alloc((size_t)cN3 * 4);
  unsigned short* ws0T = (unsigned short*)alloc((size_t)cHID * cIN * 2);
  unsigned short* wn0T = (unsigned short*)alloc((size_t)cHID * cIN * 2);
  unsigned short* ws1T = (unsigned short*)alloc((size_t)cHID * cHID * 2);
  unsigned short* wn1T = (unsigned short*)alloc((size_t)cHID * cHID * 2);
  unsigned short* ws2T = (unsigned short*)alloc((size_t)cOUTP * cHID * 2);
  unsigned short* wn2T = (unsigned short*)alloc((size_t)cOUTP * cHID * 2);
  (void)ws_size; (void)in_sizes; (void)n_in; (void)out_size;

  // zero accumulators (graph-capture-legal memsets)
  (void)hipMemsetAsync(agg0, 0, (size_t)cN1 * cIN  * 4, stream);
  (void)hipMemsetAsync(deg0, 0, (size_t)cN1 * 4, stream);
  (void)hipMemsetAsync(agg1, 0, (size_t)cN2 * cHID * 4, stream);
  (void)hipMemsetAsync(deg1, 0, (size_t)cN2 * 4, stream);
  (void)hipMemsetAsync(agg2, 0, (size_t)cN3 * cHID * 4, stream);
  (void)hipMemsetAsync(deg2, 0, (size_t)cN3 * 4, stream);

  // weight convert + transpose -> bf16 [N][K]
  w_transpose_bf16<<<(cHID * cIN  + 255) / 256, 256, 0, stream>>>(ws0, ws0T, cIN,  cHID, cHID);
  w_transpose_bf16<<<(cHID * cIN  + 255) / 256, 256, 0, stream>>>(wn0, wn0T, cIN,  cHID, cHID);
  w_transpose_bf16<<<(cHID * cHID + 255) / 256, 256, 0, stream>>>(ws1, ws1T, cHID, cHID, cHID);
  w_transpose_bf16<<<(cHID * cHID + 255) / 256, 256, 0, stream>>>(wn1, wn1T, cHID, cHID, cHID);
  w_transpose_bf16<<<(cOUTP * cHID + 255) / 256, 256, 0, stream>>>(ws2, ws2T, cHID, cOUT, cOUTP);
  w_transpose_bf16<<<(cOUTP * cHID + 255) / 256, 256, 0, stream>>>(wn2, wn2T, cHID, cOUT, cOUTP);

  const int AGG_BLOCKS = 4096;

  // ---- layer 0 ----
  sage_aggregate<false><<<AGG_BLOCKS, 256, 0, stream>>>(x, src0, dst0, agg0, deg0,
                                                        cN1 * cFAN, cIN);
  sage_gemm<false, true, cIN, cHID / 16><<<cN1 / 16, 256, 0, stream>>>(
      x, agg0, deg0, ws0T, wn0T, b0, h1, 0, 0x9E3779B9u);

  // ---- layer 1 ----
  sage_aggregate<true><<<AGG_BLOCKS, 256, 0, stream>>>(h1, src1, dst1, agg1, deg1,
                                                       cN2 * cFAN, cHID);
  sage_gemm<true, true, cHID, cHID / 16><<<cN2 / 16, 256, 0, stream>>>(
      h1, agg1, deg1, ws1T, wn1T, b1, h2, 0, 0x85EBCA6Bu);

  // ---- layer 2 (final, f32 output 8192 x 47) ----
  sage_aggregate<true><<<AGG_BLOCKS, 256, 0, stream>>>(h2, src2, dst2, agg2, deg2,
                                                       cN3 * cFAN, cHID);
  sage_gemm<true, false, cHID, cOUTP / 16><<<cN3 / 16, 256, 0, stream>>>(
      h2, agg2, deg2, ws2T, wn2T, b2, d_out, cOUT, 0u);
}
